// Qwrky7TimeMix_5617817224049
// MI455X (gfx1250) — matrix operand order repaired and checked by reading, not compiled, not run on hardware
//
#include <hip/hip_runtime.h>
#include <hip/hip_bf16.h>
#include <math.h>

// ---------------------------------------------------------------------------
// Types for CDNA5 WMMA / TDM
// ---------------------------------------------------------------------------
typedef __attribute__((ext_vector_type(16))) __bf16 v16bf;
typedef __attribute__((ext_vector_type(8)))  float  v8f;
typedef unsigned int u32x4 __attribute__((ext_vector_type(4)));
typedef int          i32x4 __attribute__((ext_vector_type(4)));
typedef int          i32x8 __attribute__((ext_vector_type(8)));

#ifdef __has_builtin
#if __has_builtin(__builtin_amdgcn_tensor_load_to_lds)
#define HAVE_TDM 1
#endif
#if __has_builtin(__builtin_amdgcn_s_wait_tensorcnt)
#define HAVE_TENSORCNT 1
#endif
#endif

// fp32 -> bf16 (round-to-nearest-even)
__device__ __forceinline__ unsigned short f2bf(float f) {
    unsigned int u = __float_as_uint(f);
    unsigned int r = u + 0x7FFFu + ((u >> 16) & 1u);
    return (unsigned short)(r >> 16);
}

// ---------------------------------------------------------------------------
// Elementwise conversion kernels
// ---------------------------------------------------------------------------
__global__ void k_f32_to_bf16(const float* __restrict__ src,
                              unsigned short* __restrict__ dst, int n) {
    int i = blockIdx.x * blockDim.x + threadIdx.x;
    if (i < n) dst[i] = f2bf(src[i]);
}

// src is [R, Cc] row-major f32; dst is [Cc, R] row-major bf16 (transpose)
__global__ void k_f32_to_bf16_T(const float* __restrict__ src,
                                unsigned short* __restrict__ dst, int R, int Cc) {
    int i = blockIdx.x * blockDim.x + threadIdx.x;
    if (i < R * Cc) {
        int r = i / Cc, c = i % Cc;
        dst[(size_t)c * R + r] = f2bf(src[i]);
    }
}

__global__ void k_tanh_bf16(const float* __restrict__ src,
                            unsigned short* __restrict__ dst, int n) {
    int i = blockIdx.x * blockDim.x + threadIdx.x;
    if (i < n) dst[i] = f2bf(tanhf(src[i]));
}

__global__ void k_sigmoid_bf16(const float* __restrict__ src,
                               unsigned short* __restrict__ dst, int n) {
    int i = blockIdx.x * blockDim.x + threadIdx.x;
    if (i < n) dst[i] = f2bf(1.0f / (1.0f + expf(-src[i])));
}

__global__ void k_copy_f32(const float* __restrict__ src,
                           float* __restrict__ dst, int n) {
    int i = blockIdx.x * blockDim.x + threadIdx.x;
    if (i < n) dst[i] = src[i];
}

// ---------------------------------------------------------------------------
// TDM helper: issue a 2D tile load (bf16 elements) global -> LDS with the
// LDS row padding folded into the descriptor (16-dword interval, 4-dword pad
// => 80-byte LDS row stride = LDSW halves). Rows/cols beyond the remaining
// tensor extent are written as zeros by the TDM.
// ---------------------------------------------------------------------------
#define LDSW 40   // padded LDS row stride in halves (80 B, multiple of 16 B)

#if defined(HAVE_TDM)
__device__ __forceinline__ void tdm_load_tile_bf16(
    const unsigned short* gtile,   // global address of tile start
    unsigned lds_addr,             // LDS byte address of tile start
    unsigned rows_rem,             // remaining rows   (tensor_dim1)
    unsigned cols_rem,             // remaining elems  (tensor_dim0)
    unsigned row_stride,           // row stride in elements (tensor_dim0_stride)
    unsigned tile_rows,            // tile_dim1
    unsigned tile_cols)            // tile_dim0
{
    unsigned long long ga = (unsigned long long)(uintptr_t)gtile;

    u32x4 g0;
    g0[0] = 1u;                                            // count=1, no gather
    g0[1] = lds_addr;                                      // lds_addr
    g0[2] = (unsigned)(ga & 0xFFFFFFFFu);                  // global_addr[31:0]
    g0[3] = (unsigned)((ga >> 32) & 0x01FFFFFFu)           // global_addr[56:32]
          | (2u << 30);                                    // type = 2 (image)

    i32x8 g1;
    // data_size=1 (2 B), pad_enable, pad_interval=3 (16 dwords),
    // pad_amount=3 (4 dwords)
    g1[0] = (int)((1u << 16) | (1u << 20) | (3u << 22) | (3u << 25));
    g1[1] = (int)((cols_rem & 0xFFFFu) << 16);             // tensor_dim0[15:0]
    g1[2] = (int)(((cols_rem >> 16) & 0xFFFFu)             // tensor_dim0[31:16]
          | ((rows_rem & 0xFFFFu) << 16));                 // tensor_dim1[15:0]
    g1[3] = (int)(((rows_rem >> 16) & 0xFFFFu)             // tensor_dim1[31:16]
          | ((tile_cols & 0xFFFFu) << 16));                // tile_dim0
    g1[4] = (int)(tile_rows & 0xFFFFu);                    // tile_dim1 (dim2=0)
    g1[5] = (int)row_stride;                               // dim0_stride[31:0]
    g1[6] = 0;                                             // stride hi / dim1_stride lo
    g1[7] = 0;                                             // dim1_stride hi

    i32x4 z4 = {0, 0, 0, 0};
#if __clang_major__ >= 23
    i32x8 z8 = {0, 0, 0, 0, 0, 0, 0, 0};
    __builtin_amdgcn_tensor_load_to_lds(g0, g1, z4, z4, z8, 0);
#else
    __builtin_amdgcn_tensor_load_to_lds(g0, g1, z4, z4, 0);
#endif
}
#endif

__device__ __forceinline__ void wait_tensor0() {
#if defined(HAVE_TENSORCNT)
    __builtin_amdgcn_s_wait_tensorcnt(0);
#else
    asm volatile("s_wait_tensorcnt 0x0" ::: "memory");
#endif
}

// ---------------------------------------------------------------------------
// Generic bf16 WMMA GEMM:  C[M,N] (f32) = A[M,K] (bf16) * B[N,K]^T (bf16)
// Block = 128 threads (4 waves). Block tile 64x64, each wave a 32x32 quadrant
// built from 2x2 v_wmma_f32_16x16x32_bf16 fragments. K is a multiple of 32
// for every GEMM here (K in {1024, 160, 64, 32}).
//
// TDM path: double-buffered software pipeline. Wave 0 issues both tile loads
// for slice ik+1 right after the top-of-iteration barrier while all waves run
// WMMA on slice ik's buffer; per-wave in-order TENSORcnt means a single
// s_wait_tensorcnt 0 + one barrier per K-slice publishes the data.
// ---------------------------------------------------------------------------
__global__ __launch_bounds__(128)
void k_gemm_bf16(const unsigned short* __restrict__ A,
                 const unsigned short* __restrict__ Bm,
                 float* __restrict__ Cm,
                 int M, int N, int K) {
#if defined(HAVE_TDM)
    __shared__ unsigned short As[2][64 * LDSW];
    __shared__ unsigned short Bs[2][64 * LDSW];
#else
    __shared__ unsigned short As[1][64 * LDSW];
    __shared__ unsigned short Bs[1][64 * LDSW];
#endif

    const int tid  = threadIdx.x;
    const int lane = tid & 31;
    const int wave = tid >> 5;
    const int m0 = blockIdx.y * 64;
    const int n0 = blockIdx.x * 64;
    const int wr = (wave >> 1) * 32;   // wave row offset in block tile
    const int wc = (wave & 1)  * 32;   // wave col offset in block tile

    v8f acc[2][2];
    #pragma unroll
    for (int i = 0; i < 2; ++i)
        #pragma unroll
        for (int j = 0; j < 2; ++j)
            acc[i][j] = v8f{0.f, 0.f, 0.f, 0.f, 0.f, 0.f, 0.f, 0.f};

    const int l15 = lane & 15;
    const int hiA = (lane < 16) ? 0 : 8;    // A: lanes16-31 hold K=8..15 / 24..31
    const int hiB = (lane < 16) ? 0 : 8;    // B: lanes16-31 hold K=8..15 / 24..31
    const int nk = K >> 5;                  // number of 32-wide K slices

#if defined(HAVE_TDM)
    // ---- pipeline prologue: slice 0 into buffer 0 ----
    if (wave == 0) {
        tdm_load_tile_bf16(A + (size_t)m0 * K,
                           (unsigned)(uintptr_t)(void*)&As[0][0],
                           (unsigned)(M - m0), (unsigned)K,
                           (unsigned)K, 64u, 32u);
        tdm_load_tile_bf16(Bm + (size_t)n0 * K,
                           (unsigned)(uintptr_t)(void*)&Bs[0][0],
                           (unsigned)(N - n0), (unsigned)K,
                           (unsigned)K, 64u, 32u);
    }
#endif

    for (int ik = 0; ik < nk; ++ik) {
        const int kk = ik << 5;
#if defined(HAVE_TDM)
        const int cur = ik & 1;
        // current slice's pair is the only outstanding TDM work for wave 0
        if (wave == 0) wait_tensor0();
        // TDM writes LDS behind the compiler's back: pin the tiles.
        asm volatile("" ::: "memory");
        __syncthreads();   // publish buf[cur]; frees buf[cur^1] for overwrite
        if (wave == 0 && ik + 1 < nk) {
            // overlap: fetch slice ik+1 while everyone computes on slice ik
            tdm_load_tile_bf16(A + (size_t)m0 * K + kk + 32,
                               (unsigned)(uintptr_t)(void*)&As[cur ^ 1][0],
                               (unsigned)(M - m0), (unsigned)(K - kk - 32),
                               (unsigned)K, 64u, 32u);
            tdm_load_tile_bf16(Bm + (size_t)n0 * K + kk + 32,
                               (unsigned)(uintptr_t)(void*)&Bs[cur ^ 1][0],
                               (unsigned)(N - n0), (unsigned)(K - kk - 32),
                               (unsigned)K, 64u, 32u);
        }
        const unsigned short* Ab = &As[cur][0];
        const unsigned short* Bb = &Bs[cur][0];
#else
        // ---- fallback: manual staging through VGPRs ----
        #pragma unroll
        for (int rep = 0; rep < 2; ++rep) {
            int cid = tid + rep * 128;      // 256 chunks = 64 rows * 4 chunks
            int row = cid >> 2;
            int ch  = cid & 3;
            uint4 val = make_uint4(0u, 0u, 0u, 0u);
            int gr = m0 + row;
            if (gr < M)
                val = *(const uint4*)(A + (size_t)gr * K + kk + ch * 8);
            *(uint4*)(&As[0][row * LDSW + ch * 8]) = val;
        }
        #pragma unroll
        for (int rep = 0; rep < 2; ++rep) {
            int cid = tid + rep * 128;
            int row = cid >> 2;
            int ch  = cid & 3;
            uint4 val = make_uint4(0u, 0u, 0u, 0u);
            int gn = n0 + row;
            if (gn < N)
                val = *(const uint4*)(Bm + (size_t)gn * K + kk + ch * 8);
            *(uint4*)(&Bs[0][row * LDSW + ch * 8]) = val;
        }
        if (kk + 32 < K) {
            int gr = m0 + (tid >> 1);
            if (gr < M)
                __builtin_prefetch(A + (size_t)gr * K + kk + 32, 0, 1);
        }
        __syncthreads();
        const unsigned short* Ab = &As[0][0];
        const unsigned short* Bb = &Bs[0][0];
#endif

        union FragU { uint4 u[2]; v16bf v; };
        FragU afr[2], bfr[2];
        #pragma unroll
        for (int i = 0; i < 2; ++i) {
            // A fragment: lane holds K = {hiA..hiA+7, 16+hiA..16+hiA+7}
            int row = wr + i * 16 + l15;
            afr[i].u[0] = *(const uint4*)(&Ab[row * LDSW + hiA]);
            afr[i].u[1] = *(const uint4*)(&Ab[row * LDSW + 16 + hiA]);
            // B fragment: lane holds K = {hiB..hiB+7, 16+hiB..16+hiB+7} of column (wc+i*16+l15)
            int col = wc + i * 16 + l15;
            bfr[i].u[0] = *(const uint4*)(&Bb[col * LDSW + hiB]);
            bfr[i].u[1] = *(const uint4*)(&Bb[col * LDSW + 16 + hiB]);
        }

        #pragma unroll
        for (int i = 0; i < 2; ++i)
            #pragma unroll
            for (int j = 0; j < 2; ++j)
                acc[i][j] = __builtin_amdgcn_wmma_f32_16x16x32_bf16(
                    false, afr[i].v, false, bfr[j].v,
                    (short)0, acc[i][j], false, false);

#if !defined(HAVE_TDM)
        __syncthreads();   // single-buffer fallback needs the trailing barrier
#endif
    }

    // ---- store D: lane L -> col = l15, rows = r + (L<16 ? 0 : 8) ----
    const int rbase = (lane < 16) ? 0 : 8;
    #pragma unroll
    for (int i = 0; i < 2; ++i) {
        #pragma unroll
        for (int j = 0; j < 2; ++j) {
            int col = n0 + wc + j * 16 + l15;
            if (col < N) {
                int rowt = m0 + wr + i * 16 + rbase;
                #pragma unroll
                for (int r = 0; r < 8; ++r) {
                    int row = rowt + r;
                    if (row < M)
                        Cm[(size_t)row * N + col] = acc[i][j][r];
                }
            }
        }
    }
}

// ---------------------------------------------------------------------------
// Stage 2: per-(b,t,h) fused prep. 64 threads per head slice.
// ---------------------------------------------------------------------------
__global__ __launch_bounds__(64)
void k_stage2(const float* __restrict__ katt, const float* __restrict__ vatt,
              float* __restrict__ wlog_decay, float* __restrict__ apre_k,
              float* __restrict__ vpre_v, float* __restrict__ abuf,
              float* __restrict__ bbuf, const float* __restrict__ vfirst,
              const float* __restrict__ w0, const float* __restrict__ a0,
              const float* __restrict__ v0, const float* __restrict__ k_k,
              const float* __restrict__ k_a) {
    __shared__ float red[64];
    int bid  = blockIdx.x;           // m*H + h,  m = b*T + t
    int h    = bid & 15;
    int m    = bid >> 4;
    int lane = threadIdx.x;
    int c    = h * 64 + lane;
    int cat  = (h >> 1) * 64 + lane;  // GQA: GROUP=2
    size_t idx  = (size_t)m * 1024 + c;
    size_t idxa = (size_t)m * 512 + cat;

    float kx = katt[idxa];
    float vx = vatt[idxa];
    float dec = expf(-expf(w0[c] + wlog_decay[idx]));
    float iclr = 1.0f / (1.0f + expf(-(a0[c] + apre_k[idx])));
    float eff  = 1.0f + (iclr - 1.0f) * k_a[c];
    float vs   = 1.0f / (1.0f + expf(-(v0[c] + vpre_v[idx])));
    float vv   = vx + (vfirst[idx] - vx) * vs;

    float kkv = kx * k_k[c];
    red[lane] = kkv * kkv;
    __syncthreads();
    #pragma unroll
    for (int s = 32; s > 0; s >>= 1) {
        if (lane < s) red[lane] += red[lane + s];
        __syncthreads();
    }
    float nrm = sqrtf(red[0]);
    float kkn = kkv / fmaxf(nrm, 1e-12f);

    wlog_decay[idx] = dec;
    apre_k[idx]     = kx * eff;
    vpre_v[idx]     = vv;
    abuf[idx]       = -kkn;
    bbuf[idx]       = kkn * eff;
}

// ---------------------------------------------------------------------------
// Stage 3: WKV scan. One 64-thread block per (b,h); thread i owns state row i
// (64 floats in registers). Operands broadcast through LDS each step.
// ---------------------------------------------------------------------------
__global__ __launch_bounds__(64)
void k_scan(const float* __restrict__ rbuf, const float* __restrict__ decay,
            const float* __restrict__ kbuf, const float* __restrict__ vbuf,
            const float* __restrict__ abuf, const float* __restrict__ bbuf,
            const float* __restrict__ S0, float* __restrict__ xx,
            float* __restrict__ S_out) {
    int bh = blockIdx.x;             // b*H + h
    int b  = bh >> 4;
    int h  = bh & 15;
    int i  = threadIdx.x;            // state row

    __shared__ float rs[64], ws[64], ks[64], as[64], bs[64];

    float S[64];
    const float* S0p = S0 + (size_t)bh * 4096 + (size_t)i * 64;
    #pragma unroll
    for (int j = 0; j < 64; ++j) S[j] = S0p[j];

    size_t base = (size_t)b * 1024 * 1024 + (size_t)h * 64;
    for (int t = 0; t < 1024; ++t) {
        size_t off = base + (size_t)t * 1024;
        float vi = vbuf[off + i];
        rs[i] = rbuf[off + i];
        ws[i] = decay[off + i];
        ks[i] = kbuf[off + i];
        as[i] = abuf[off + i];
        bs[i] = bbuf[off + i];
        __syncthreads();

        float sa = 0.f;
        #pragma unroll
        for (int j = 0; j < 64; ++j) sa += S[j] * as[j];
        float y = 0.f;
        #pragma unroll
        for (int j = 0; j < 64; ++j) {
            S[j] = S[j] * ws[j] + sa * bs[j] + vi * ks[j];
            y += S[j] * rs[j];
        }
        xx[off + i] = y;
        __syncthreads();
    }

    float* Sop = S_out + (size_t)bh * 4096 + (size_t)i * 64;
    #pragma unroll
    for (int j = 0; j < 64; ++j) Sop[j] = S[j];
}

// ---------------------------------------------------------------------------
// Stage 4: per-head GroupNorm + r_k bonus + gate -> bf16 operand for Wo GEMM
// ---------------------------------------------------------------------------
__global__ __launch_bounds__(64)
void k_epilogue(const float* __restrict__ xx, const float* __restrict__ rbuf,
                const float* __restrict__ kbuf, const float* __restrict__ vbuf,
                const float* __restrict__ g, const float* __restrict__ r_k,
                const float* __restrict__ ln_w, const float* __restrict__ ln_b,
                unsigned short* __restrict__ xngb) {
    __shared__ float red[64], red2[64];
    int bid  = blockIdx.x;           // m*H + h
    int h    = bid & 15;
    int m    = bid >> 4;
    int lane = threadIdx.x;
    int c    = h * 64 + lane;
    size_t idx = (size_t)m * 1024 + c;

    float x = xx[idx];
    red[lane]  = x;
    red2[lane] = x * x;
    __syncthreads();
    #pragma unroll
    for (int s = 32; s > 0; s >>= 1) {
        if (lane < s) { red[lane] += red[lane + s]; red2[lane] += red2[lane + s]; }
        __syncthreads();
    }
    float mu  = red[0] * (1.0f / 64.0f);
    float var = fmaxf(red2[0] * (1.0f / 64.0f) - mu * mu, 0.0f);
    float xn  = (x - mu) * rsqrtf(var + 1e-5f * 64.0f);
    xn = xn * ln_w[c] + ln_b[c];
    __syncthreads();

    red[lane] = rbuf[idx] * kbuf[idx] * r_k[h * 64 + lane];
    __syncthreads();
    #pragma unroll
    for (int s = 32; s > 0; s >>= 1) {
        if (lane < s) red[lane] += red[lane + s];
        __syncthreads();
    }
    xn += red[0] * vbuf[idx];
    xngb[idx] = f2bf(xn * g[idx]);
}

// ---------------------------------------------------------------------------
// Host-side orchestration
// ---------------------------------------------------------------------------
extern "C" void kernel_launch(void* const* d_in, const int* in_sizes, int n_in,
                              void* d_out, int out_size, void* d_ws, size_t ws_size,
                              hipStream_t stream) {
    (void)in_sizes; (void)n_in; (void)out_size; (void)ws_size;
    const int B = 2, T = 1024, C = 1024, CA = 512, H = 16;
    const int M = B * T;   // 2048

    const float* x      = (const float*)d_in[0];
    const float* S0     = (const float*)d_in[1];
    const float* vfirst = (const float*)d_in[2];
    const float* w0     = (const float*)d_in[3];
    const float* w1     = (const float*)d_in[4];
    const float* w2     = (const float*)d_in[5];
    const float* a0     = (const float*)d_in[6];
    const float* a1     = (const float*)d_in[7];
    const float* a2     = (const float*)d_in[8];
    const float* v0     = (const float*)d_in[9];
    const float* v1     = (const float*)d_in[10];
    const float* v2     = (const float*)d_in[11];
    const float* g1     = (const float*)d_in[12];
    const float* g2     = (const float*)d_in[13];
    const float* k_k    = (const float*)d_in[14];
    const float* k_a    = (const float*)d_in[15];
    const float* r_k    = (const float*)d_in[16];
    const float* Wq     = (const float*)d_in[17];
    const float* Wk     = (const float*)d_in[18];
    const float* Wv     = (const float*)d_in[19];
    const float* Wo     = (const float*)d_in[20];
    const float* ln_w   = (const float*)d_in[21];
    const float* ln_b   = (const float*)d_in[22];

    float* outp  = (float*)d_out;                 // [B,T,C]
    float* Soutp = outp + (size_t)M * C;          // [B,H,64,64]
    float* vfo   = Soutp + (size_t)B * H * 64 * 64;

    // --- bump allocator over d_ws ---
    char* ws = (char*)d_ws;
    size_t off = 0;
    auto alloc = [&](size_t bytes) -> void* {
        void* p = ws + off;
        off += (bytes + 255) & ~(size_t)255;
        return p;
    };

    unsigned short* xbf  = (unsigned short*)alloc((size_t)M * C * 2);
    unsigned short* wqbf = (unsigned short*)alloc((size_t)C * C * 2);
    unsigned short* wkbf = (unsigned short*)alloc((size_t)CA * C * 2);
    unsigned short* wvbf = (unsigned short*)alloc((size_t)CA * C * 2);
    unsigned short* wobf = (unsigned short*)alloc((size_t)C * C * 2);
    unsigned short* w1T  = (unsigned short*)alloc((size_t)64 * C * 2);
    unsigned short* w2T  = (unsigned short*)alloc((size_t)C * 64 * 2);
    unsigned short* a1T  = (unsigned short*)alloc((size_t)64 * C * 2);
    unsigned short* a2T  = (unsigned short*)alloc((size_t)C * 64 * 2);
    unsigned short* v1T  = (unsigned short*)alloc((size_t)32 * C * 2);
    unsigned short* v2T  = (unsigned short*)alloc((size_t)C * 32 * 2);
    unsigned short* g1T  = (unsigned short*)alloc((size_t)160 * C * 2);
    unsigned short* g2T  = (unsigned short*)alloc((size_t)C * 160 * 2);

    float* rbuf = (float*)alloc((size_t)M * C * 4);
    float* katt = (float*)alloc((size_t)M * CA * 4);
    float* vatt = (float*)alloc((size_t)M * CA * 4);
    float* hw   = (float*)alloc((size_t)M * 64 * 4);
    float* ha   = (float*)alloc((size_t)M * 64 * 4);
    float* hv   = (float*)alloc((size_t)M * 32 * 4);
    float* hg   = (float*)alloc((size_t)M * 160 * 4);
    unsigned short* hwb = (unsigned short*)alloc((size_t)M * 64 * 2);
    unsigned short* hab = (unsigned short*)alloc((size_t)M * 64 * 2);
    unsigned short* hvb = (unsigned short*)alloc((size_t)M * 32 * 2);
    unsigned short* hgb = (unsigned short*)alloc((size_t)M * 160 * 2);
    float* wlog = (float*)alloc((size_t)M * C * 4);   // becomes decay
    float* apre = (float*)alloc((size_t)M * C * 4);   // becomes k
    float* vpre = (float*)alloc((size_t)M * C * 4);   // becomes v
    float* gbuf = (float*)alloc((size_t)M * C * 4);
    float* abuf = (float*)alloc((size_t)M * C * 4);
    float* bbuf = (float*)alloc((size_t)M * C * 4);
    float* xxb  = (float*)alloc((size_t)M * C * 4);
    unsigned short* xngb = (unsigned short*)alloc((size_t)M * C * 2);

    auto grid1 = [](int n) { return dim3((n + 255) / 256); };

    // ---- Stage 0: conversions ----
    k_f32_to_bf16<<<grid1(M * C), 256, 0, stream>>>(x, xbf, M * C);
    k_f32_to_bf16<<<grid1(C * C), 256, 0, stream>>>(Wq, wqbf, C * C);
    k_f32_to_bf16<<<grid1(CA * C), 256, 0, stream>>>(Wk, wkbf, CA * C);
    k_f32_to_bf16<<<grid1(CA * C), 256, 0, stream>>>(Wv, wvbf, CA * C);
    k_f32_to_bf16<<<grid1(C * C), 256, 0, stream>>>(Wo, wobf, C * C);
    k_f32_to_bf16_T<<<grid1(C * 64), 256, 0, stream>>>(w1, w1T, C, 64);
    k_f32_to_bf16_T<<<grid1(64 * C), 256, 0, stream>>>(w2, w2T, 64, C);
    k_f32_to_bf16_T<<<grid1(C * 64), 256, 0, stream>>>(a1, a1T, C, 64);
    k_f32_to_bf16_T<<<grid1(64 * C), 256, 0, stream>>>(a2, a2T, 64, C);
    k_f32_to_bf16_T<<<grid1(C * 32), 256, 0, stream>>>(v1, v1T, C, 32);
    k_f32_to_bf16_T<<<grid1(32 * C), 256, 0, stream>>>(v2, v2T, 32, C);
    k_f32_to_bf16_T<<<grid1(C * 160), 256, 0, stream>>>(g1, g1T, C, 160);
    k_f32_to_bf16_T<<<grid1(160 * C), 256, 0, stream>>>(g2, g2T, 160, C);

    auto gemm = [&](const unsigned short* A, const unsigned short* Bm, float* Cm,
                    int m, int n, int k) {
        dim3 grid((n + 63) / 64, (m + 63) / 64);
        k_gemm_bf16<<<grid, 128, 0, stream>>>(A, Bm, Cm, m, n, k);
    };

    // ---- Stage 1: projection GEMMs (all WMMA bf16) ----
    gemm(xbf, wqbf, rbuf, M, C, C);     // r     = x @ Wq^T
    gemm(xbf, wkbf, katt, M, CA, C);    // k_att = x @ Wk^T
    gemm(xbf, wvbf, vatt, M, CA, C);    // v_att = x @ Wv^T
    gemm(xbf, w1T, hw, M, 64, C);       // x @ w1
    gemm(xbf, a1T, ha, M, 64, C);       // x @ a1
    gemm(xbf, v1T, hv, M, 32, C);       // x @ v1
    gemm(xbf, g1T, hg, M, 160, C);      // x @ g1

    k_tanh_bf16<<<grid1(M * 64), 256, 0, stream>>>(hw, hwb, M * 64);
    k_f32_to_bf16<<<grid1(M * 64), 256, 0, stream>>>(ha, hab, M * 64);
    k_f32_to_bf16<<<grid1(M * 32), 256, 0, stream>>>(hv, hvb, M * 32);
    k_sigmoid_bf16<<<grid1(M * 160), 256, 0, stream>>>(hg, hgb, M * 160);

    gemm(hwb, w2T, wlog, M, C, 64);     // tanh(x@w1) @ w2
    gemm(hab, a2T, apre, M, C, 64);     // (x@a1) @ a2
    gemm(hvb, v2T, vpre, M, C, 32);     // (x@v1) @ v2
    gemm(hgb, g2T, gbuf, M, C, 160);    // sigmoid(x@g1) @ g2

    // ---- Stage 2: fused per-head prep ----
    k_stage2<<<dim3(M * H), 64, 0, stream>>>(katt, vatt, wlog, apre, vpre,
                                             abuf, bbuf, vfirst, w0, a0, v0,
                                             k_k, k_a);

    // ---- Stage 3: sequential WKV scan ----
    k_scan<<<dim3(B * H), 64, 0, stream>>>(rbuf, wlog, apre, vpre, abuf, bbuf,
                                           S0, xxb, Soutp);

    // ---- Stage 4: GroupNorm + bonus + gate -> bf16 ----
    k_epilogue<<<dim3(M * H), 64, 0, stream>>>(xxb, rbuf, apre, vpre, gbuf,
                                               r_k, ln_w, ln_b, xngb);

    // ---- Stage 5: output projection ----
    gemm(xngb, wobf, outp, M, C, C);    // out = (xn*g) @ Wo^T

    // ---- Stage 6: v_first passthrough ----
    k_copy_f32<<<grid1(M * C), 256, 0, stream>>>(vfirst, vfo, M * C);
}
